// GATv2ConvOrdered_88175678587728
// MI455X (gfx1250) — matrix operand order repaired and checked by reading, not compiled, not run on hardware
//
#include <hip/hip_runtime.h>

// ---------------- problem constants (from reference) ----------------
#define NN     20000   // nodes
#define DD     16      // neighbors incl self-loop
#define NF     256     // NFEAT
#define F1C    256     // NHEADS*NHID
#define H1C    256     // OUTD2
#define G1C    1024    // 4*OUTD2
#define K1C    512     // F1 + OUTD2 (concat GEMM K)
#define C2C    32      // NCLASS
#define G2C    128     // 4*NCLASS
#define K2C    64      // NCLASS + NCLASS

typedef __bf16 bf16;
typedef __attribute__((ext_vector_type(16))) __bf16 v16bf;
typedef __attribute__((ext_vector_type(8)))  __bf16 v8bf;
typedef __attribute__((ext_vector_type(8)))  float  v8f;

__device__ __forceinline__ v8f vzero8() {
  v8f z = {0.f,0.f,0.f,0.f,0.f,0.f,0.f,0.f};
  return z;
}

// A-fragment (16x32 bf16, row-major source with given element stride).
// Layout: lanes 0-15 (M=lane): K 0..15 ; lanes 16-31 (M=lane-16): K 16..31 (matches B fragment).
__device__ __forceinline__ v16bf load_afrag(const bf16* A, int stride, int m, int half, int kbase) {
  const bf16* p = A + m * stride + kbase + half * 16;
  v8bf lo = *(const v8bf*)(p);
  v8bf hi = *(const v8bf*)(p + 8);
  return __builtin_shufflevector(lo, hi, 0,1,2,3,4,5,6,7,8,9,10,11,12,13,14,15);
}

// B-fragment (32x16 bf16) from column-major weight W_t[col][K]:
// lane holds column (lane&15); lanes 0-15: K kbase..kbase+15 ; lanes 16-31: K kbase+16..kbase+31.
__device__ __forceinline__ v16bf load_bfrag(const bf16* Wt, int Kstride, int col, int half, int kbase) {
  return *(const v16bf*)(Wt + col * Kstride + kbase + half * 16);
}

__device__ __forceinline__ v8f wmma_bf16(v16bf a, v16bf b, v8f c) {
  return __builtin_amdgcn_wmma_f32_16x16x32_bf16(false, a, false, b, (short)0, c, false, false);
}

__device__ __forceinline__ float sigf(float x) { return 1.f / (1.f + __expf(-x)); }

// ---------------- K0a: generic pack: out[NC][KA+KB] = concat(A,B)^T in bf16 ----------------
__global__ __launch_bounds__(256) void k_pack(const float* __restrict__ A, const float* __restrict__ B,
                                              int KA, int KB, int NC, bf16* __restrict__ out, int total) {
  int idx = blockIdx.x * 256 + threadIdx.x;
  if (idx >= total) return;
  int Ktot = KA + KB;
  int j = idx / Ktot, k = idx - j * Ktot;
  float v = (k < KA) ? A[k * NC + j] : B[(k - KA) * NC + j];
  out[idx] = (bf16)v;
}

// K0b: LSTM-1 gate-permuted pack. Packed col j -> (wave w=j>>7, tile nt=(j&127)>>4, lane n=j&15),
// gate = nt&3, kchunk = nt>>2, k = 32w + kchunk*16 + n, original column = gate*256 + k.
// Each half of a wave's 8 tiles = all 4 gates for one k-chunk -> 64-VGPR accumulator passes.
__global__ __launch_bounds__(256) void k_pack_l1(const float* __restrict__ Wx, const float* __restrict__ Wh,
                                                 bf16* __restrict__ out) {
  int idx = blockIdx.x * 256 + threadIdx.x;            // total = 1024*512
  int j = idx >> 9, kk = idx & 511;
  int w = j >> 7, rem = j & 127, nt = rem >> 4, n = rem & 15;
  int gate = nt & 3;
  int k = 32 * w + (nt >> 2) * 16 + n;
  int oc = gate * H1C + k;
  float v = (kk < F1C) ? Wx[kk * G1C + oc] : Wh[(kk - F1C) * G1C + oc];
  out[idx] = (bf16)v;
}

// K0c: LSTM-2 gate-permuted pack. Packed col j -> (w=j>>6, nt=(j&63)>>4, n=j&15),
// gate = nt, k = 16w + n, original column = gate*32 + k.
__global__ __launch_bounds__(256) void k_pack_l2(const float* __restrict__ Wx, const float* __restrict__ Wh,
                                                 bf16* __restrict__ out) {
  int idx = blockIdx.x * 256 + threadIdx.x;            // total = 128*64
  if (idx >= G2C * K2C) return;
  int j = idx >> 6, kk = idx & 63;
  int w = j >> 6, rem = j & 63, nt = rem >> 4, n = rem & 15;
  int k = 16 * w + n;
  int oc = nt * C2C + k;
  float v = (kk < C2C) ? Wx[kk * G2C + oc] : Wh[(kk - C2C) * G2C + oc];
  out[idx] = (bf16)v;
}

// ---------------- K1: xl = x@Wl1, xr = x@Wr1 (K=256, Ncols=256 each) ----------------
__global__ __launch_bounds__(256) void k_gemm1(const float* __restrict__ x,
                                               const bf16* __restrict__ WlT, const bf16* __restrict__ WrT,
                                               float* __restrict__ xl, float* __restrict__ xr) {
  __shared__ bf16 As[16 * NF];                       // 8 KB
  const int row0 = blockIdx.x * 16;
  const int t = threadIdx.x;
  {
    int r = t >> 4, c0 = (t & 15) * 16;
    const float* s = x + (size_t)(row0 + r) * NF + c0;
    bf16* d = As + r * NF + c0;
#pragma unroll
    for (int i = 0; i < 16; ++i) d[i] = (bf16)s[i];
  }
  __syncthreads();
  const int w = t >> 5, lane = t & 31, half = lane >> 4, mn = lane & 15;
  const bf16* WT = (w < 4) ? WlT : WrT;
  float* out = (w < 4) ? xl : xr;
  const int colbase = (w & 3) * 64;
  v8f acc[4]; for (int i = 0; i < 4; ++i) acc[i] = vzero8();
  for (int kt = 0; kt < 8; ++kt) {
    v16bf a = load_afrag(As, NF, mn, half, kt * 32);
#pragma unroll
    for (int nt = 0; nt < 4; ++nt) {
      v16bf b = load_bfrag(WT, NF, colbase + nt * 16 + mn, half, kt * 32);
      acc[nt] = wmma_bf16(a, b, acc[nt]);
    }
  }
#pragma unroll
  for (int nt = 0; nt < 4; ++nt)
#pragma unroll
    for (int r = 0; r < 8; ++r)
      out[(size_t)(row0 + r + 8 * half) * F1C + colbase + nt * 16 + mn] = acc[nt][r];
}

// ---------------- K2: GATv2 layer-1 attention (online softmax), one wave per node --------------
__global__ __launch_bounds__(256) void k_att1(const float* __restrict__ xl, const float* __restrict__ xr,
                                              const int* __restrict__ src,
                                              const float* __restrict__ a1, const float* __restrict__ b1,
                                              float* __restrict__ h1,
                                              int* __restrict__ pidx, float* __restrict__ pw) {
  const int node = blockIdx.x * 8 + (threadIdx.x >> 5);
  const int lane = threadIdx.x & 31;                 // lane covers cols 8*lane..8*lane+7, head = lane>>2
  float xrv[8], av[8], outv[8];
#pragma unroll
  for (int i = 0; i < 8; ++i) {
    xrv[i] = xr[(size_t)node * F1C + 8 * lane + i];
    av[i]  = a1[8 * lane + i];
    outv[i] = 0.f;
  }
  float e[16]; int idxs[16];
  float mrun = -1e30f, srun = 0.f;
  for (int d = 0; d < DD; ++d) {
    int idx = src[node * DD + d]; idxs[d] = idx;
    const float* nbp = xl + (size_t)idx * F1C + 8 * lane;
    float nbv[8];
#pragma unroll
    for (int i = 0; i < 8; ++i) nbv[i] = nbp[i];
    float s = 0.f;
#pragma unroll
    for (int i = 0; i < 8; ++i) {
      float v = nbv[i] + xrv[i];
      v = v > 0.f ? v : 0.2f * v;                    // leaky_relu(0.2)
      s += av[i] * v;
    }
    s += __shfl_xor(s, 1, 32);                       // reduce over 4-lane head group
    s += __shfl_xor(s, 2, 32);
    e[d] = s;
    float mnew = fmaxf(mrun, s);
    float scale = __expf(mrun - mnew);
    float p = __expf(s - mnew);
    srun = srun * scale + p;
#pragma unroll
    for (int i = 0; i < 8; ++i) outv[i] = outv[i] * scale + p * nbv[i];
    mrun = mnew;
  }
  float inv = 1.f / srun;
#pragma unroll
  for (int i = 0; i < 8; ++i)
    h1[(size_t)node * F1C + 8 * lane + i] = outv[i] * inv + b1[8 * lane + i];
  // head-mean attention: all 4 lanes per head hold identical att; xor4/8/16 sums 8 distinct heads
  float attm[16];
  for (int d = 0; d < DD; ++d) {
    float a = __expf(e[d] - mrun) * inv;
    a += __shfl_xor(a, 4, 32);
    a += __shfl_xor(a, 8, 32);
    a += __shfl_xor(a, 16, 32);
    attm[d] = a * 0.125f;
  }
  if (lane == 0) {
    int ord[16];
#pragma unroll
    for (int i = 0; i < 16; ++i) ord[i] = i;
    for (int i = 1; i < 16; ++i) {                   // stable insertion sort, ascending
      int o = ord[i]; float v = attm[o]; int j = i - 1;
      while (j >= 0 && attm[ord[j]] > v) { ord[j + 1] = ord[j]; --j; }
      ord[j + 1] = o;
    }
    for (int tt = 0; tt < 16; ++tt) {
      pidx[node * DD + tt] = idxs[ord[tt]];
      pw[node * DD + tt]   = attm[ord[tt]];
    }
  }
}

// ---------------- K3: LSTM aggregation 1, M=32, two gate-half passes (64-VGPR accumulators) ----
// 32 nodes/block, 16 steps. LDS = A operand only (32 KB). Each B-fragment feeds two M-tiles and
// is read exactly once per step; gates are folded pass-by-pass so only 4 N-tiles are live at once.
__global__ __launch_bounds__(256) void k_lstm1(const float* __restrict__ h1,
                                               const int* __restrict__ pidx, const float* __restrict__ pw,
                                               const bf16* __restrict__ WcT, const float* __restrict__ bl1,
                                               float* __restrict__ h2) {
  __shared__ bf16 As[32 * K1C];                      // [xt | h] bf16, 32 KB
  const int node0 = blockIdx.x * 32;
  const int t = threadIdx.x, w = t >> 5, lane = t & 31, half = lane >> 4, mn = lane & 15;
  for (int i = t; i < 32 * H1C; i += 256)
    As[(i >> 8) * K1C + F1C + (i & 255)] = (bf16)0.f;  // h = 0
  const int colbase = w * 128;
  // per-lane bias for (gate, kchunk): k = 32w + hk*16 + n
  float bi[2], bff[2], bg[2], bo[2];
#pragma unroll
  for (int hk = 0; hk < 2; ++hk) {
    int k = 32 * w + hk * 16 + mn;
    bi[hk]  = bl1[k];
    bff[hk] = bl1[H1C + k];
    bg[hk]  = bl1[2 * H1C + k];
    bo[hk]  = bl1[3 * H1C + k];
  }
  float creg[2][2][8];                                // cell state [hk][mtile][r]
  float hbuf[2][2][8];                                // deferred h output [hk][mtile][r]
#pragma unroll
  for (int hk = 0; hk < 2; ++hk)
#pragma unroll
    for (int mt = 0; mt < 2; ++mt)
#pragma unroll
      for (int r = 0; r < 8; ++r) creg[hk][mt][r] = 0.f;

  for (int step = 0; step < 16; ++step) {
    { // stage xt = h1[perm_idx] * perm_w  (bf16); 256 threads x 32 elems over 32 rows
      int r = t >> 3, c0 = (t & 7) * 32;
      int gidx = pidx[(node0 + r) * DD + step];
      float wt = pw[(node0 + r) * DD + step];
      const float* s = h1 + (size_t)gidx * F1C + c0;
      bf16* dst = As + r * K1C + c0;
#pragma unroll
      for (int i = 0; i < 32; ++i) dst[i] = (bf16)(s[i] * wt);
    }
    __syncthreads();
    for (int hk = 0; hk < 2; ++hk) {                  // two gate-half passes
      v8f acc[2][4];
#pragma unroll
      for (int mt = 0; mt < 2; ++mt)
        for (int i = 0; i < 4; ++i) acc[mt][i] = vzero8();
      for (int kt = 0; kt < 16; ++kt) {
        v16bf a0 = load_afrag(As,            K1C, mn, half, kt * 32);
        v16bf a1 = load_afrag(As + 16 * K1C, K1C, mn, half, kt * 32);
#pragma unroll
        for (int nt = 0; nt < 4; ++nt) {
          v16bf b = load_bfrag(WcT, K1C, colbase + (hk * 4 + nt) * 16 + mn, half, kt * 32);
          acc[0][nt] = wmma_bf16(a0, b, acc[0][nt]);
          acc[1][nt] = wmma_bf16(a1, b, acc[1][nt]);
        }
      }
      // fold gates for this k-chunk: acc[mt][0..3] = i, f, g, o
#pragma unroll
      for (int mt = 0; mt < 2; ++mt)
#pragma unroll
        for (int r = 0; r < 8; ++r) {
          float gi = acc[mt][0][r] + bi[hk];
          float gf = acc[mt][1][r] + bff[hk];
          float gg = acc[mt][2][r] + bg[hk];
          float go = acc[mt][3][r] + bo[hk];
          float c = sigf(gf) * creg[hk][mt][r] + sigf(gi) * tanhf(gg);
          float h = sigf(go) * tanhf(c);
          creg[hk][mt][r] = c;
          hbuf[hk][mt][r] = h;
        }
    }
    __syncthreads();                                  // all A reads (both passes) done
#pragma unroll
    for (int hk = 0; hk < 2; ++hk) {
      int k = 32 * w + hk * 16 + mn;
#pragma unroll
      for (int mt = 0; mt < 2; ++mt)
#pragma unroll
        for (int r = 0; r < 8; ++r) {
          float h = hbuf[hk][mt][r];
          int row = mt * 16 + r + 8 * half;
          As[row * K1C + F1C + k] = (bf16)h;
          if (step == 15) {
            float eo = h > 0.f ? h : (__expf(h) - 1.f);  // fused ELU
            h2[(size_t)(node0 + row) * H1C + k] = eo;
          }
        }
    }
    __syncthreads();                                  // h writes visible before next GEMM
  }
}

// ---------------- K4: xl2 = h2@Wl2, xr2 = h2@Wr2 (K=256, Ncols=32) ----------------
__global__ __launch_bounds__(64) void k_gemm2(const float* __restrict__ h2,
                                              const bf16* __restrict__ Wl2T, const bf16* __restrict__ Wr2T,
                                              float* __restrict__ xl2, float* __restrict__ xr2) {
  __shared__ bf16 As[16 * H1C];                      // 8 KB
  const int row0 = blockIdx.x * 16;
  const int t = threadIdx.x;
  {
    int r = t >> 2, c0 = (t & 3) * 64;
    const float* s = h2 + (size_t)(row0 + r) * H1C + c0;
    bf16* d = As + r * H1C + c0;
#pragma unroll
    for (int i = 0; i < 64; ++i) d[i] = (bf16)s[i];
  }
  __syncthreads();
  const int w = t >> 5, lane = t & 31, half = lane >> 4, mn = lane & 15;
  const bf16* WT = (w == 0) ? Wl2T : Wr2T;
  float* out = (w == 0) ? xl2 : xr2;
  v8f acc[2]; acc[0] = vzero8(); acc[1] = vzero8();
  for (int kt = 0; kt < 8; ++kt) {
    v16bf a = load_afrag(As, H1C, mn, half, kt * 32);
#pragma unroll
    for (int nt = 0; nt < 2; ++nt) {
      v16bf b = load_bfrag(WT, H1C, nt * 16 + mn, half, kt * 32);
      acc[nt] = wmma_bf16(a, b, acc[nt]);
    }
  }
#pragma unroll
  for (int nt = 0; nt < 2; ++nt)
#pragma unroll
    for (int r = 0; r < 8; ++r)
      out[(size_t)(row0 + r + 8 * half) * C2C + nt * 16 + mn] = acc[nt][r];
}

// ---------------- K5: GATv2 layer-2 attention (1 head, 32 cols), one wave per node -------------
__global__ __launch_bounds__(256) void k_att2(const float* __restrict__ xl2, const float* __restrict__ xr2,
                                              const int* __restrict__ src,
                                              const float* __restrict__ a2, const float* __restrict__ b2,
                                              float* __restrict__ h3,
                                              int* __restrict__ pidx2, float* __restrict__ pw2) {
  const int node = blockIdx.x * 8 + (threadIdx.x >> 5);
  const int lane = threadIdx.x & 31;                 // lane == class column
  float xrv = xr2[(size_t)node * C2C + lane];
  float av  = a2[lane];
  float e[16]; int idxs[16];
  float mrun = -1e30f, srun = 0.f, outv = 0.f;
  for (int d = 0; d < DD; ++d) {
    int idx = src[node * DD + d]; idxs[d] = idx;
    float nb = xl2[(size_t)idx * C2C + lane];
    float v = nb + xrv;
    v = v > 0.f ? v : 0.2f * v;
    float s = av * v;
#pragma unroll
    for (int off = 1; off < 32; off <<= 1) s += __shfl_xor(s, off, 32);
    e[d] = s;
    float mnew = fmaxf(mrun, s);
    float scale = __expf(mrun - mnew);
    float p = __expf(s - mnew);
    srun = srun * scale + p;
    outv = outv * scale + p * nb;
    mrun = mnew;
  }
  float inv = 1.f / srun;
  h3[(size_t)node * C2C + lane] = outv * inv + b2[lane];
  if (lane == 0) {
    float attm[16]; int ord[16];
    for (int d = 0; d < DD; ++d) { attm[d] = __expf(e[d] - mrun) * inv; ord[d] = d; }
    for (int i = 1; i < 16; ++i) {
      int o = ord[i]; float v = attm[o]; int j = i - 1;
      while (j >= 0 && attm[ord[j]] > v) { ord[j + 1] = ord[j]; --j; }
      ord[j + 1] = o;
    }
    for (int tt = 0; tt < 16; ++tt) {
      pidx2[node * DD + tt] = idxs[ord[tt]];
      pw2[node * DD + tt]   = attm[ord[tt]];
    }
  }
}

// ---------------- K6: LSTM aggregation 2 (register-resident gates) + log_softmax ----------------
// 2 waves, 16 nodes/block. Wave w owns gate tiles nt=gate, k = 16w + n.
__global__ __launch_bounds__(64) void k_lstm2(const float* __restrict__ h3,
                                              const int* __restrict__ pidx, const float* __restrict__ pw,
                                              const bf16* __restrict__ Wc2T, const float* __restrict__ bl2,
                                              float* __restrict__ out) {
  __shared__ bf16  As[16 * K2C];                     // 2 KB
  __shared__ bf16  Ws[G2C * K2C];                    // 16 KB, weights resident across steps
  __shared__ float Hs[16 * C2C];                     // 2 KB, final hidden for log_softmax
  const int node0 = blockIdx.x * 16;
  const int t = threadIdx.x, w = t >> 5, lane = t & 31, half = lane >> 4, mn = lane & 15;
  for (int i = t; i < G2C * K2C; i += 64) Ws[i] = Wc2T[i];
  for (int i = t; i < 16 * C2C; i += 64)
    As[(i >> 5) * K2C + C2C + (i & 31)] = (bf16)0.f;
  const int colbase = w * 64;
  const int kcol = 16 * w + mn;
  float bi = bl2[kcol], bff = bl2[C2C + kcol], bg = bl2[2 * C2C + kcol], bo = bl2[3 * C2C + kcol];
  float creg[8];
#pragma unroll
  for (int r = 0; r < 8; ++r) creg[r] = 0.f;
  __syncthreads();
  for (int step = 0; step < 16; ++step) {
    { // stage xt: 64 threads x 8 elems
      int r = t >> 2, c0 = (t & 3) * 8;
      int gidx = pidx[(node0 + r) * DD + step];
      float wt = pw[(node0 + r) * DD + step];
#pragma unroll
      for (int i = 0; i < 8; ++i)
        As[r * K2C + c0 + i] = (bf16)(h3[(size_t)gidx * C2C + c0 + i] * wt);
    }
    __syncthreads();
    v8f acc[4]; for (int i = 0; i < 4; ++i) acc[i] = vzero8();
#pragma unroll
    for (int kt = 0; kt < 2; ++kt) {
      v16bf a = load_afrag(As, K2C, mn, half, kt * 32);
#pragma unroll
      for (int nt = 0; nt < 4; ++nt) {
        v16bf b = load_bfrag(Ws, K2C, colbase + nt * 16 + mn, half, kt * 32);
        acc[nt] = wmma_bf16(a, b, acc[nt]);
      }
    }
    __syncthreads();
#pragma unroll
    for (int r = 0; r < 8; ++r) {
      float gi = acc[0][r] + bi;
      float gf = acc[1][r] + bff;
      float gg = acc[2][r] + bg;
      float go = acc[3][r] + bo;
      float c = sigf(gf) * creg[r] + sigf(gi) * tanhf(gg);
      float h = sigf(go) * tanhf(c);
      creg[r] = c;
      int row = r + 8 * half;
      As[row * K2C + C2C + kcol] = (bf16)h;
      if (step == 15) Hs[row * C2C + kcol] = h;
    }
    __syncthreads();
  }
  if (t < 16) {
    int r = t;
    float mx = -1e30f;
    for (int c = 0; c < C2C; ++c) mx = fmaxf(mx, Hs[r * C2C + c]);
    float s = 0.f;
    for (int c = 0; c < C2C; ++c) s += __expf(Hs[r * C2C + c] - mx);
    float ls = mx + __logf(s);
    for (int c = 0; c < C2C; ++c)
      out[(size_t)(node0 + r) * C2C + c] = Hs[r * C2C + c] - ls;
  }
}

// ---------------- host side ----------------
extern "C" void kernel_launch(void* const* d_in, const int* in_sizes, int n_in,
                              void* d_out, int out_size, void* d_ws, size_t ws_size,
                              hipStream_t stream) {
  const float* x   = (const float*)d_in[0];
  const int*   src = (const int*)  d_in[1];
  const float* Wl1 = (const float*)d_in[2];
  const float* Wr1 = (const float*)d_in[3];
  const float* a1  = (const float*)d_in[4];
  const float* b1  = (const float*)d_in[5];
  const float* Wx1 = (const float*)d_in[6];
  const float* Wh1 = (const float*)d_in[7];
  const float* bl1 = (const float*)d_in[8];
  const float* Wl2 = (const float*)d_in[9];
  const float* Wr2 = (const float*)d_in[10];
  const float* a2  = (const float*)d_in[11];
  const float* b2  = (const float*)d_in[12];
  const float* Wx2 = (const float*)d_in[13];
  const float* Wh2 = (const float*)d_in[14];
  const float* bl2 = (const float*)d_in[15];
  float* outp = (float*)d_out;

  char* p = (char*)d_ws;
  auto take = [&](size_t bytes) -> char* {
    char* r = p;
    p += (bytes + 255) & ~(size_t)255;
    return r;
  };
  bf16* Wl1T = (bf16*)take((size_t)F1C * NF * 2);
  bf16* Wr1T = (bf16*)take((size_t)F1C * NF * 2);
  bf16* Wc1T = (bf16*)take((size_t)G1C * K1C * 2);
  bf16* Wl2T = (bf16*)take((size_t)C2C * H1C * 2);
  bf16* Wr2T = (bf16*)take((size_t)C2C * H1C * 2);
  bf16* Wc2T = (bf16*)take((size_t)G2C * K2C * 2);
  float* xl   = (float*)take((size_t)NN * F1C * 4);
  float* xr   = (float*)take((size_t)NN * F1C * 4);
  float* h1   = (float*)take((size_t)NN * F1C * 4);
  float* h2   = (float*)take((size_t)NN * H1C * 4);
  float* xl2  = (float*)take((size_t)NN * C2C * 4);
  float* xr2  = (float*)take((size_t)NN * C2C * 4);
  float* h3   = (float*)take((size_t)NN * C2C * 4);
  int*   pidx  = (int*)  take((size_t)NN * DD * 4);
  float* pwgt  = (float*)take((size_t)NN * DD * 4);
  int*   pidx2 = (int*)  take((size_t)NN * DD * 4);
  float* pwgt2 = (float*)take((size_t)NN * DD * 4);

  auto packs = [&](const float* A, const float* B, int KA, int KB, int NC, bf16* o) {
    int total = NC * (KA + KB);
    k_pack<<<(total + 255) / 256, 256, 0, stream>>>(A, B, KA, KB, NC, o, total);
  };
  packs(Wl1, Wl1, NF, 0, F1C, Wl1T);       // [256][256] -> colmajor bf16
  packs(Wr1, Wr1, NF, 0, F1C, Wr1T);
  k_pack_l1<<<(G1C * K1C) / 256, 256, 0, stream>>>(Wx1, Wh1, Wc1T);   // gate-permuted [1024][512]
  packs(Wl2, Wl2, H1C, 0, C2C, Wl2T);
  packs(Wr2, Wr2, H1C, 0, C2C, Wr2T);
  k_pack_l2<<<(G2C * K2C + 255) / 256, 256, 0, stream>>>(Wx2, Wh2, Wc2T);  // gate-permuted [128][64]

  k_gemm1<<<NN / 16, 256, 0, stream>>>(x, Wl1T, Wr1T, xl, xr);
  k_att1 <<<NN / 8, 256, 0, stream>>>(xl, xr, src, a1, b1, h1, pidx, pwgt);
  k_lstm1<<<NN / 32, 256, 0, stream>>>(h1, pidx, pwgt, Wc1T, bl1, h2);
  k_gemm2<<<NN / 16, 64, 0, stream>>>(h2, Wl2T, Wr2T, xl2, xr2);
  k_att2 <<<NN / 8, 256, 0, stream>>>(xl2, xr2, src, a2, b2, h3, pidx2, pwgt2);
  k_lstm2<<<NN / 16, 64, 0, stream>>>(h3, pidx2, pwgt2, Wc2T, bl2, outp);
}
